// HGCN_18949395710344
// MI455X (gfx1250) — compile-verified, matrix operand order checked by reading, not run on hardware
//
#include <hip/hip_runtime.h>
#include <hip/hip_bf16.h>
#include <stdint.h>

// ---------------- problem dimensions ----------------
#define PP    65536   // H*W pixels
#define NSUP  4096
#define CIN   200
#define CPAD  208     // 13*16 ; col 200 holds the implicit "ones" column (colsum)
#define HID   128
#define C2    64
#define NCLS  16
#define XPANEL (CPAD * 32)   // 6656 u16 per 32-pixel panel, fragment order

typedef __attribute__((ext_vector_type(16))) __bf16          v16bf;
typedef __attribute__((ext_vector_type(16))) unsigned short  v16u;
typedef __attribute__((ext_vector_type(8)))  unsigned short  u16x8;
typedef __attribute__((ext_vector_type(8)))  float           v8f;

static __device__ inline v8f v8f_zero() {
  v8f z = {0.f, 0.f, 0.f, 0.f, 0.f, 0.f, 0.f, 0.f};
  return z;
}

// float -> bf16 bits, round-to-nearest-even
static __device__ inline unsigned short f2bf(float f) {
  unsigned int u = __float_as_uint(f);
  u += 0x7FFFu + ((u >> 16) & 1u);
  return (unsigned short)(u >> 16);
}

// ---------------- fragment-order layout ----------------
// WMMA 16x16x32 bf16 operand layout (A rows / B cols both = "rc"):
//   lane = (rc&15) + 16*((k>>3)&1) ;  element e = (k&7) + 8*(k>>4)
// Tiles are stored (in LDS and in the pre-converted X buffer) so that each
// lane's 16 elements are CONTIGUOUS (32 bytes):
//   fragment load = 2x ds_load_b128, no packing VALU.
// A tile of R rows x 32 K = (R/16) blocks of 512 u16.

static __device__ inline v16bf frag_sw(const unsigned short* t, int lane) {
  const v16u* p = (const v16u*)(t + lane * 16);
  return __builtin_bit_cast(v16bf, *p);
}

// scatter one element into fragment order (source contiguous in rc)
static __device__ inline void store_one(unsigned short* tile, int rc, int k, float f) {
  int blk = rc >> 4, rcl = rc & 15;
  int lane = rcl + (((k >> 3) & 1) << 4);
  int e = (k & 7) + ((k >> 4) << 3);
  tile[blk * 512 + lane * 16 + e] = f2bf(f);
}

// store one 8-wide k-octet (16 contiguous LDS bytes -> ds_store_b128)
static __device__ inline void store_oct(unsigned short* tile, int rc, int o, const float* f) {
  int blk = rc >> 4, rcl = rc & 15;
  int lane = rcl + ((o & 1) << 4);
  int e0 = (o >> 1) << 3;
  u16x8 v;
#pragma unroll
  for (int e = 0; e < 8; ++e) v[e] = f2bf(f[e]);
  *(u16x8*)(tile + blk * 512 + lane * 16 + e0) = v;
}

// 8 contiguous floats via two b128 loads
static __device__ inline void load8(const float* p, float* f) {
  const float4* s = (const float4*)p;
  float4 x0 = s[0], x1 = s[1];
  f[0] = x0.x; f[1] = x0.y; f[2] = x0.z; f[3] = x0.w;
  f[4] = x1.x; f[5] = x1.y; f[6] = x1.z; f[7] = x1.w;
}

#define WMMA_BF16(a, b, c) \
  __builtin_amdgcn_wmma_f32_16x16x32_bf16(false, (a), false, (b), (short)0, (c), false, false)

// ---------------- kernels ----------------

__global__ void k_zero(float* __restrict__ p, int n) {
  int i = blockIdx.x * 256 + threadIdx.x;
  if (i < n) p[i] = 0.0f;
}

// One-time: convert [X | 1 | 0pad] into bf16 fragment-panel order.
// Panel pp covers pixels p0=32*pp; within: 13 blocks of 512 u16 (LDS-identical).
// grid = PP/32 blocks, 256 threads.
__global__ void __launch_bounds__(256)
k_xpanel(const float* __restrict__ X, unsigned short* __restrict__ XP) {
  const int pp = blockIdx.x, p0 = pp * 32;
  unsigned short* dst = XP + (size_t)pp * XPANEL;
  for (int idx = threadIdx.x; idx < 32 * CPAD; idx += 256) {
    int k = idx / CPAD, c = idx % CPAD;   // coalesced reads along c
    float v = (c < CIN) ? X[(size_t)(p0 + k) * CIN + c] : (c == CIN ? 1.0f : 0.0f);
    store_one(dst, c, k, v);
  }
}

// G = Q^T @ [X | 1 | 0pad]   -> [NSUP, CPAD], split-K over P with fp32 atomics.
// block = 256 (8 waves), grid = (NSUP/128, 16)
__global__ void __launch_bounds__(256)
k_qtx(const float* __restrict__ Q, const unsigned short* __restrict__ XP,
      float* __restrict__ G) {
  const int MT = 128, KT = 32, SPLITK = 16, KCH = PP / SPLITK;
  __shared__ alignas(32) unsigned short a_lds[(MT / 16) * 512];    // Q^T tile, frag order
  __shared__ alignas(32) unsigned short b_lds[(CPAD / 16) * 512];  // X panel, frag order
  const int i0 = blockIdx.x * MT;
  const int p_begin = blockIdx.y * KCH;
  const int t = threadIdx.x, lane = t & 31, m0 = (t >> 5) * 16;

  v8f acc[13];
#pragma unroll
  for (int j = 0; j < 13; ++j) acc[j] = v8f_zero();

  for (int p0 = p_begin; p0 < p_begin + KCH; p0 += KT) {
    __syncthreads();
    // A: Q[p0+k][i0+m] -- global contiguous in m (coalesced), swizzled b16 scatter
#pragma unroll 4
    for (int idx = t; idx < KT * MT; idx += 256) {
      int k = idx / MT, m = idx % MT;
      store_one(a_lds, m, k, Q[(size_t)(p0 + k) * NSUP + i0 + m]);
    }
    // B: pre-converted X panel -- pure b128 copy, no conversion VALU
    {
      const u16x8* src = (const u16x8*)(XP + (size_t)(p0 >> 5) * XPANEL);
      u16x8* dst = (u16x8*)b_lds;
#pragma unroll 4
      for (int idx = t; idx < XPANEL / 8; idx += 256) dst[idx] = src[idx];
    }
    if (p0 + KT < p_begin + KCH)
      __builtin_prefetch(&Q[(size_t)(p0 + KT + lane) * NSUP + i0], 0, 0);
    __syncthreads();
    v16bf a = frag_sw(a_lds + (m0 >> 4) * 512, lane);
#pragma unroll
    for (int j = 0; j < 13; ++j) {
      v16bf b = frag_sw(b_lds + j * 512, lane);
      acc[j] = WMMA_BF16(a, b, acc[j]);
    }
  }
  const int n = lane & 15, g = lane >> 4;
#pragma unroll
  for (int j = 0; j < 13; ++j)
#pragma unroll
    for (int v = 0; v < 8; ++v)
      atomicAdd(&G[(size_t)(i0 + m0 + v + 8 * g) * CPAD + j * 16 + n], acc[j][v]);
}

// H1 = (G[:, :200] / G[:,200]) @ W1 + b1   -> [NSUP, HID]
// block = 128, grid = NSUP/64
__global__ void __launch_bounds__(128)
k_scale_w1(const float* __restrict__ G, const float* __restrict__ W1,
           const float* __restrict__ b1, float* __restrict__ H1) {
  const int MT = 64, KT = 32;
  __shared__ float inv_lds[MT];
  __shared__ alignas(32) unsigned short a_lds[(MT / 16) * 512];
  __shared__ alignas(32) unsigned short b_lds[(HID / 16) * 512];
  const int i0 = blockIdx.x * MT;
  const int t = threadIdx.x, lane = t & 31, m0 = (t >> 5) * 16;
  if (t < MT) inv_lds[t] = 1.0f / G[(size_t)(i0 + t) * CPAD + CIN];
  v8f acc[8];
#pragma unroll
  for (int j = 0; j < 8; ++j) acc[j] = v8f_zero();
  __syncthreads();
  for (int k0 = 0; k0 < CPAD; k0 += KT) {       // 7 steps, zero-padded past 200
    __syncthreads();
    // A: scaled G rows, k-contiguous -> octet path
    for (int idx = t; idx < MT * 4; idx += 128) {
      int m = idx >> 2, o = idx & 3;
      float f[8];
      load8(G + (size_t)(i0 + m) * CPAD + k0 + o * 8, f);
      float s = inv_lds[m];
#pragma unroll
      for (int e = 0; e < 8; ++e) f[e] = (k0 + o * 8 + e < CIN) ? f[e] * s : 0.0f;
      store_oct(a_lds, m, o, f);
    }
    // B: W1 rows contiguous in h
    for (int idx = t; idx < KT * HID; idx += 128) {
      int k = idx / HID, h = idx % HID, c = k0 + k;
      store_one(b_lds, h, k, (c < CIN) ? W1[(size_t)c * HID + h] : 0.0f);
    }
    __syncthreads();
    v16bf a = frag_sw(a_lds + (m0 >> 4) * 512, lane);
#pragma unroll
    for (int j = 0; j < 8; ++j) {
      v16bf b = frag_sw(b_lds + j * 512, lane);
      acc[j] = WMMA_BF16(a, b, acc[j]);
    }
  }
  const int n = lane & 15, g = lane >> 4;
#pragma unroll
  for (int j = 0; j < 8; ++j)
#pragma unroll
    for (int v = 0; v < 8; ++v)
      H1[(size_t)(i0 + m0 + v + 8 * g) * HID + j * 16 + n] = acc[j][v] + b1[j * 16 + n];
}

// Sout += Ahyper @ Hsrc   (NCOLS = 128 or 64), split-K over NSUP with atomics.
// block = 128, grid = (NSUP/64, 8)
template <int NCOLS>
__global__ void __launch_bounds__(128)
k_aprop(const float* __restrict__ Ah, const float* __restrict__ Hsrc, float* __restrict__ Sout) {
  const int MT = 64, KT = 32, SPLITK = 8, KCH = NSUP / SPLITK, NT = NCOLS / 16;
  __shared__ alignas(32) unsigned short a_lds[(MT / 16) * 512];
  __shared__ alignas(32) unsigned short b_lds[NT * 512];
  const int i0 = blockIdx.x * MT;
  const int k_begin = blockIdx.y * KCH;
  const int t = threadIdx.x, lane = t & 31, m0 = (t >> 5) * 16;
  v8f acc[NT];
#pragma unroll
  for (int j = 0; j < NT; ++j) acc[j] = v8f_zero();
  for (int k0 = k_begin; k0 < k_begin + KCH; k0 += KT) {
    __syncthreads();
    // A row-major, k-contiguous -> octet path (b128 global + b128 LDS)
    for (int idx = t; idx < MT * 4; idx += 128) {
      int m = idx >> 2, o = idx & 3;
      float f[8];
      load8(Ah + (size_t)(i0 + m) * NSUP + k0 + o * 8, f);
      store_oct(a_lds, m, o, f);
    }
    // B: H rows contiguous in n
    for (int idx = t; idx < KT * NCOLS; idx += 128) {
      int k = idx / NCOLS, n = idx % NCOLS;
      store_one(b_lds, n, k, Hsrc[(size_t)(k0 + k) * NCOLS + n]);
    }
    __syncthreads();
    v16bf a = frag_sw(a_lds + (m0 >> 4) * 512, lane);
#pragma unroll
    for (int j = 0; j < NT; ++j) {
      v16bf b = frag_sw(b_lds + j * 512, lane);
      acc[j] = WMMA_BF16(a, b, acc[j]);
    }
  }
  const int n = lane & 15, g = lane >> 4;
#pragma unroll
  for (int j = 0; j < NT; ++j)
#pragma unroll
    for (int v = 0; v < 8; ++v)
      atomicAdd(&Sout[(size_t)(i0 + m0 + v + 8 * g) * NCOLS + j * 16 + n], acc[j][v]);
}

// per-column mean & inv-std over NSUP rows (biased variance, training-mode BN)
__global__ void __launch_bounds__(256)
k_bnstats(const float* __restrict__ S, float* __restrict__ stats, int ncols) {
  __shared__ float r1[256], r2[256];
  const int c = blockIdx.x, tid = threadIdx.x;
  float s = 0.f, s2 = 0.f;
  for (int i = tid; i < NSUP; i += 256) {
    float v = S[(size_t)i * ncols + c];
    s += v; s2 += v * v;
  }
  r1[tid] = s; r2[tid] = s2;
  __syncthreads();
  for (int off = 128; off > 0; off >>= 1) {
    if (tid < off) { r1[tid] += r1[tid + off]; r2[tid] += r2[tid + off]; }
    __syncthreads();
  }
  if (tid == 0) {
    float mean = r1[0] / (float)NSUP;
    float var  = r2[0] / (float)NSUP - mean * mean;
    stats[c]         = mean;
    stats[ncols + c] = rsqrtf(var + 1e-5f);
  }
}

__global__ void __launch_bounds__(256)
k_bnapply(const float* __restrict__ S, const float* __restrict__ stats,
          const float* __restrict__ gma, const float* __restrict__ bta,
          float* __restrict__ R, int ncols) {
  int idx = blockIdx.x * 256 + threadIdx.x;
  if (idx < NSUP * ncols) {
    int c = idx % ncols;
    float v = (S[idx] - stats[c]) * stats[ncols + c] * gma[c] + bta[c];
    R[idx] = v > 0.f ? v : 0.f;
  }
}

// H2 = R1 @ W2 + b2  -> [NSUP, 64]; block = 128, grid = NSUP/64
__global__ void __launch_bounds__(128)
k_w2(const float* __restrict__ R1, const float* __restrict__ W2,
     const float* __restrict__ b2, float* __restrict__ H2) {
  const int MT = 64, KT = 32, NT = C2 / 16;
  __shared__ alignas(32) unsigned short a_lds[(MT / 16) * 512];
  __shared__ alignas(32) unsigned short b_lds[NT * 512];
  const int i0 = blockIdx.x * MT;
  const int t = threadIdx.x, lane = t & 31, m0 = (t >> 5) * 16;
  v8f acc[NT];
#pragma unroll
  for (int j = 0; j < NT; ++j) acc[j] = v8f_zero();
  for (int k0 = 0; k0 < HID; k0 += KT) {
    __syncthreads();
    for (int idx = t; idx < MT * 4; idx += 128) {   // R1 k-contiguous
      int m = idx >> 2, o = idx & 3;
      float f[8];
      load8(R1 + (size_t)(i0 + m) * HID + k0 + o * 8, f);
      store_oct(a_lds, m, o, f);
    }
    for (int idx = t; idx < KT * C2; idx += 128) {  // W2 contiguous in n
      int k = idx / C2, n = idx % C2;
      store_one(b_lds, n, k, W2[(size_t)(k0 + k) * C2 + n]);
    }
    __syncthreads();
    v16bf a = frag_sw(a_lds + (m0 >> 4) * 512, lane);
#pragma unroll
    for (int j = 0; j < NT; ++j) {
      v16bf b = frag_sw(b_lds + j * 512, lane);
      acc[j] = WMMA_BF16(a, b, acc[j]);
    }
  }
  const int n = lane & 15, g = lane >> 4;
#pragma unroll
  for (int j = 0; j < NT; ++j)
#pragma unroll
    for (int v = 0; v < 8; ++v)
      H2[(size_t)(i0 + m0 + v + 8 * g) * C2 + j * 16 + n] = acc[j][v] + b2[j * 16 + n];
}

// Z = R2 @ lin_w^T + lin_b  (tiny: K=64, N=16) — plain VALU
__global__ void __launch_bounds__(256)
k_lin(const float* __restrict__ R2, const float* __restrict__ lw,
      const float* __restrict__ lb, float* __restrict__ Z) {
  int idx = blockIdx.x * 256 + threadIdx.x;
  if (idx < NSUP * NCLS) {
    int i = idx / NCLS, cl = idx % NCLS;
    float s = lb[cl];
#pragma unroll
    for (int h = 0; h < C2; ++h) s += R2[(size_t)i * C2 + h] * lw[cl * C2 + h];
    Z[idx] = s;
  }
}

// out = softmax(Q @ Z) ; block = 256 (8 waves x 16 rows), grid = PP/128
__global__ void __launch_bounds__(256)
k_qz_softmax(const float* __restrict__ Q, const float* __restrict__ Z, float* __restrict__ out) {
  const int MT = 128, KT = 32;
  __shared__ alignas(32) unsigned short a_lds[(MT / 16) * 512];
  __shared__ alignas(32) unsigned short b_lds[512];
  const int r0 = blockIdx.x * MT;
  const int t = threadIdx.x, lane = t & 31, m0 = (t >> 5) * 16;
  v8f acc = v8f_zero();
  for (int k0 = 0; k0 < NSUP; k0 += KT) {
    __syncthreads();
    // Q row-major, k-contiguous -> octet path
#pragma unroll 2
    for (int idx = t; idx < MT * 4; idx += 256) {
      int m = idx >> 2, o = idx & 3;
      float f[8];
      load8(Q + (size_t)(r0 + m) * NSUP + k0 + o * 8, f);
      store_oct(a_lds, m, o, f);
    }
    for (int idx = t; idx < KT * NCLS; idx += 256) {
      int k = idx / NCLS, n = idx % NCLS;
      store_one(b_lds, n, k, Z[(size_t)(k0 + k) * NCLS + n]);
    }
    if (k0 + KT < NSUP)
      __builtin_prefetch(&Q[(size_t)(r0 + (t >> 1)) * NSUP + k0 + KT], 0, 0);
    __syncthreads();
    v16bf a = frag_sw(a_lds + (m0 >> 4) * 512, lane);
    v16bf b = frag_sw(b_lds, lane);
    acc = WMMA_BF16(a, b, acc);
  }
  // softmax across the 16 classes: classes live in the 16 lanes of each half-wave
#pragma unroll
  for (int v = 0; v < 8; ++v) {
    float x = acc[v];
    float mx = x;
#pragma unroll
    for (int off = 1; off < 16; off <<= 1) mx = fmaxf(mx, __shfl_xor(mx, off, 16));
    float e = __expf(x - mx);
    float sm = e;
#pragma unroll
    for (int off = 1; off < 16; off <<= 1) sm += __shfl_xor(sm, off, 16);
    acc[v] = e / sm;
  }
  const int n = lane & 15, g = lane >> 4;
#pragma unroll
  for (int v = 0; v < 8; ++v)
    out[(size_t)(r0 + m0 + v + 8 * g) * NCLS + n] = acc[v];
}

// ---------------- host-side launch ----------------
extern "C" void kernel_launch(void* const* d_in, const int* in_sizes, int n_in,
                              void* d_out, int out_size, void* d_ws, size_t ws_size,
                              hipStream_t stream) {
  const float* x     = (const float*)d_in[0];
  const float* Q     = (const float*)d_in[1];
  const float* A     = (const float*)d_in[2];
  const float* W1    = (const float*)d_in[3];
  const float* b1    = (const float*)d_in[4];
  const float* W2    = (const float*)d_in[5];
  const float* b2    = (const float*)d_in[6];
  const float* g2    = (const float*)d_in[7];
  const float* beta2 = (const float*)d_in[8];
  const float* g3    = (const float*)d_in[9];
  const float* beta3 = (const float*)d_in[10];
  const float* lw    = (const float*)d_in[11];
  const float* lb    = (const float*)d_in[12];

  float* ws = (float*)d_ws;
  float* G   = ws;                       // NSUP*CPAD
  float* H1  = G   + NSUP * CPAD;        // NSUP*HID
  float* S1  = H1  + NSUP * HID;         // NSUP*HID (atomics)
  float* R1  = S1  + NSUP * HID;         // NSUP*HID
  float* H2  = R1  + NSUP * HID;         // NSUP*C2
  float* S2  = H2  + NSUP * C2;          // NSUP*C2 (atomics)
  float* R2  = S2  + NSUP * C2;          // NSUP*C2
  float* Zb  = R2  + NSUP * C2;          // NSUP*NCLS
  float* st1 = Zb  + NSUP * NCLS;        // 2*HID
  float* st2 = st1 + 2 * HID;            // 2*C2
  // bf16 X panels, 32-byte aligned
  uintptr_t xp_addr = ((uintptr_t)(st2 + 2 * C2) + 31) & ~(uintptr_t)31;
  unsigned short* XP = (unsigned short*)xp_addr;   // (PP/32)*XPANEL u16 (~27 MB)

  // zero atomic-accumulated buffers (ws is poisoned, not zeroed)
  k_zero<<<(NSUP * CPAD + 255) / 256, 256, 0, stream>>>(G, NSUP * CPAD);
  k_zero<<<(NSUP * HID  + 255) / 256, 256, 0, stream>>>(S1, NSUP * HID);
  k_zero<<<(NSUP * C2   + 255) / 256, 256, 0, stream>>>(S2, NSUP * C2);

  // pre-convert [X|1] into bf16 fragment panels (one pass, coalesced reads)
  k_xpanel<<<PP / 32, 256, 0, stream>>>(x, XP);
  // G = Q^T @ [X|1], split-K=16
  k_qtx<<<dim3(NSUP / 128, 16), 256, 0, stream>>>(Q, XP, G);
  // H1 = colnorm(G) @ W1 + b1
  k_scale_w1<<<NSUP / 64, 128, 0, stream>>>(G, W1, b1, H1);
  // S1 = A @ H1
  k_aprop<HID><<<dim3(NSUP / 64, 8), 128, 0, stream>>>(A, H1, S1);
  // BN + ReLU
  k_bnstats<<<HID, 256, 0, stream>>>(S1, st1, HID);
  k_bnapply<<<(NSUP * HID + 255) / 256, 256, 0, stream>>>(S1, st1, g2, beta2, R1, HID);
  // H2 = R1 @ W2 + b2
  k_w2<<<NSUP / 64, 128, 0, stream>>>(R1, W2, b2, H2);
  // S2 = A @ H2
  k_aprop<C2><<<dim3(NSUP / 64, 8), 128, 0, stream>>>(A, H2, S2);
  // BN + ReLU
  k_bnstats<<<C2, 256, 0, stream>>>(S2, st2, C2);
  k_bnapply<<<(NSUP * C2 + 255) / 256, 256, 0, stream>>>(S2, st2, g3, beta3, R2, C2);
  // Z = R2 @ lin_w^T + lin_b
  k_lin<<<(NSUP * NCLS + 255) / 256, 256, 0, stream>>>(R2, lw, lb, Zb);
  // out = softmax(Q @ Z)
  k_qz_softmax<<<PP / 128, 256, 0, stream>>>(Q, Zb, (float*)d_out);
}
